// GridTransformerBlock_47244640256235
// MI455X (gfx1250) — compile-verified, matrix operand order checked by reading, not run on hardware
//
#include <hip/hip_runtime.h>
#include <hip/hip_bf16.h>

// ---------------------------------------------------------------------------
// GridTransformerBlock for MI455X (gfx1250), bf16 WMMA pipeline.
// B=8, S=16384, E=256, FF=1024, 512 windows of T=256 tokens.
// All matmuls via v_wmma_f32_16x16x32_bf16 (f32 accumulate).
// ---------------------------------------------------------------------------

typedef __bf16 bf16_t;
typedef __attribute__((ext_vector_type(16))) __bf16 v16bf;
typedef __attribute__((ext_vector_type(8)))  __bf16 v8bf;
typedef __attribute__((ext_vector_type(8)))  float  v8f;

__device__ __forceinline__ v8f zero8() {
  v8f z = {0.f, 0.f, 0.f, 0.f, 0.f, 0.f, 0.f, 0.f};
  return z;
}

// Load one wave32 A/B operand tile (16x32 bf16) for wmma_f32_16x16x32_bf16.
// Per ISA layout: lane L holds row (L&15), K = (L>>4)*8 + {0..7} and +{16..23}.
// p must already point at (row, kchunk_base + (L>>4)*8); both 16B-aligned loads.
__device__ __forceinline__ v16bf ldA16(const bf16_t* p) {
  v8bf lo = *(const v8bf*)(p);
  v8bf hi = *(const v8bf*)(p + 16);
  v16bf r;
#pragma unroll
  for (int i = 0; i < 8; ++i) { r[i] = lo[i]; r[i + 8] = hi[i]; }
  return r;
}

__device__ __forceinline__ v8f wmma_bf16(v16bf a, v16bf b, v8f c) {
  // (neg_a, A, neg_b, B, c_mod, C, reuse_a, reuse_b)
  return __builtin_amdgcn_wmma_f32_16x16x32_bf16(false, a, false, b, (short)0, c,
                                                 false, false);
}

// window permutation: token-major flat index of element (row g in (B*nW*T), e)
__device__ __forceinline__ long win_flat(int g, int e) {
  int b = g >> 14, r = g & 16383;
  int win = r >> 8, tt = r & 255;
  int hn = win >> 3, wn = win & 7;
  return (long)b * 4194304 + (long)tt * 16384 + hn * 2048 + wn * 16 +
         ((e >> 4) << 7) + (e & 15);
}

// ---------------------------------------------------------------------------
// Prep: weights -> bf16 transposed ([n][k], contiguous in k = B-operand ready)
// ---------------------------------------------------------------------------
__global__ void prep_weights_kernel(const float* Wq, const float* Wk, const float* Wv,
                                    const float* Wo, const float* W1, const float* W2,
                                    bf16_t* WqT, bf16_t* WkT, bf16_t* WvT, bf16_t* WoT,
                                    bf16_t* W1T, bf16_t* W2T) {
  int i = blockIdx.x * blockDim.x + threadIdx.x;  // 0..262143
  if (i < 65536) {
    int k = i >> 8, n = i & 255;
    int to = n * 256 + k, fr = k * 256 + n;
    WqT[to] = (bf16_t)Wq[fr];
    WkT[to] = (bf16_t)Wk[fr];
    WvT[to] = (bf16_t)Wv[fr];
    WoT[to] = (bf16_t)Wo[fr];
  }
  {  // W1: (E=256, FF=1024) -> W1T[n][k], n in FF
    int n = i & 1023, k = i >> 10;
    W1T[n * 256 + k] = (bf16_t)W1[k * 1024 + n];
  }
  {  // W2: (FF=1024, E=256) -> W2T[n][k], n in E, k in FF
    int k = i >> 8, n = i & 255;
    W2T[n * 1024 + k] = (bf16_t)W2[k * 256 + n];
  }
}

// gather x (B,S,E f32, window-permuted) -> tb bf16 (131072 x 256, token rows)
__global__ void gather_x_kernel(const float* x, bf16_t* tb) {
  int g = blockIdx.x;    // 0..131071
  int e = threadIdx.x;   // 0..255
  tb[(long)g * 256 + e] = (bf16_t)x[win_flat(g, e)];
}

// ---------------------------------------------------------------------------
// Generic WMMA GEMM: C(M x N) = A(M x K bf16 row-major) * BT(N x K bf16)^T
// 256 threads = 8 waves, workgroup tile 64x64, wave tile 16x32 (2 wmma/k-step)
// modes: 0 = bf16 row-major + bias      (Q, K)
//        1 = bf16 window-transposed + bias (V -> vT[win][n][t])
//        2 = f32 + bf16 scatter (window permutation) + bias (attn out-proj)
//        3 = exact GELU, bf16 row-major + bias (FFN1)
//        4 = bf16 row-major, per-window B (attn @ V), no bias
// ---------------------------------------------------------------------------
__global__ void gemm_kernel(const bf16_t* A, const bf16_t* BT, const float* bias,
                            void* out0, void* out1, int K, int N, int mode) {
  const int lane = threadIdx.x & 31;
  const int wv = threadIdx.x >> 5;      // 0..7
  const int wr = wv >> 1, wc = wv & 1;  // 4 x 2 wave grid
  const int rowBase = blockIdx.x * 64 + wr * 16;
  const int colBase = blockIdx.y * 64 + wc * 32;
  const int l15 = lane & 15;
  const int kOff = (lane >> 4) << 3;  // 0 or 8

  const bf16_t* Bt = BT;
  if (mode == 4) Bt += (long)(blockIdx.x >> 2) * 65536;  // window = rows/256

  const bf16_t* aPtr  = A  + (long)(rowBase + l15) * K + kOff;
  const bf16_t* bPtr0 = Bt + (long)(colBase + l15) * K + kOff;
  const bf16_t* bPtr1 = bPtr0 + (long)16 * K;

  v8f acc0 = zero8(), acc1 = zero8();
  for (int kc = 0; kc < K; kc += 32) {
    __builtin_prefetch(aPtr + kc + 128, 0, 1);  // global_prefetch_b8
    v16bf a  = ldA16(aPtr + kc);
    v16bf b0 = ldA16(bPtr0 + kc);
    v16bf b1 = ldA16(bPtr1 + kc);
    acc0 = wmma_bf16(a, b0, acc0);
    acc1 = wmma_bf16(a, b1, acc1);
  }

  // C/D layout: reg r -> row rb+r, col = l15 (+16 per ntile)
  const int rb = rowBase + ((lane >> 4) << 3);
  v8f accs[2] = {acc0, acc1};
#pragma unroll
  for (int nt = 0; nt < 2; ++nt) {
    int col = colBase + nt * 16 + l15;
    float bv = (mode == 4 || bias == nullptr) ? 0.f : bias[col];
#pragma unroll
    for (int r = 0; r < 8; ++r) {
      int row = rb + r;
      float v = accs[nt][r] + bv;
      if (mode == 0 || mode == 4) {
        ((bf16_t*)out0)[(long)row * N + col] = (bf16_t)v;
      } else if (mode == 1) {
        // V^T per window: vT[win*65536 + n*256 + t]
        ((bf16_t*)out0)[(long)(row >> 8) * 65536 + (long)col * 256 + (row & 255)] =
            (bf16_t)v;
      } else if (mode == 2) {
        long flat = win_flat(row, col);   // inverse of gather = same permutation
        ((float*)out0)[flat] = v;         // y (f32) for residuals
        ((bf16_t*)out1)[flat] = (bf16_t)v;  // y (bf16) as FFN GEMM input
      } else {  // mode 3: exact GELU
        float gv = 0.5f * v * (1.0f + erff(v * 0.70710678f));
        ((bf16_t*)out0)[(long)row * N + col] = (bf16_t)gv;
      }
    }
  }
}

// ---------------------------------------------------------------------------
// Fused scores + softmax, per window. Each wave owns a 16-row query strip:
// 16x256 scores in accumulators (16 ntiles x 8 kchunks = 128 wmma), then
// row softmax via 16-lane shfl_xor reductions (matches wave32 C/D halves),
// attn written bf16 row-major (ready as A operand for the AV GEMM).
// ---------------------------------------------------------------------------
__global__ void attn_softmax_kernel(const bf16_t* qb, const bf16_t* kb, bf16_t* attnb) {
  const int lane = threadIdx.x & 31;
  const int wv = threadIdx.x >> 5;
  const int rowtile = blockIdx.x * 8 + wv;  // 0..8191
  const int g0 = rowtile * 16;              // global query row base
  const int winBase = (g0 >> 8) << 8;       // window's first row
  const int l15 = lane & 15;
  const int kOff = (lane >> 4) << 3;

  const bf16_t* qrow = qb + (long)(g0 + l15) * 256 + kOff;
  const bf16_t* kwin = kb + (long)winBase * 256;

  v16bf qa[8];
#pragma unroll
  for (int kc = 0; kc < 8; ++kc) qa[kc] = ldA16(qrow + kc * 32);

  v8f acc[16];
#pragma unroll
  for (int nt = 0; nt < 16; ++nt) {
    v8f c = zero8();
    const bf16_t* bp = kwin + (long)(nt * 16 + l15) * 256 + kOff;
#pragma unroll
    for (int kc = 0; kc < 8; ++kc) c = wmma_bf16(qa[kc], ldA16(bp + kc * 32), c);
    acc[nt] = c;
  }

  const float scale = 0.0625f;  // 1/sqrt(E=256)
  const int rb = g0 + ((lane >> 4) << 3);
#pragma unroll
  for (int r = 0; r < 8; ++r) {
    float m = -3.4e38f;
#pragma unroll
    for (int nt = 0; nt < 16; ++nt) m = fmaxf(m, acc[nt][r]);
    for (int off = 1; off < 16; off <<= 1) m = fmaxf(m, __shfl_xor(m, off, 32));
    float p[16];
    float s = 0.f;
#pragma unroll
    for (int nt = 0; nt < 16; ++nt) {
      p[nt] = __expf((acc[nt][r] - m) * scale);
      s += p[nt];
    }
    for (int off = 1; off < 16; off <<= 1) s += __shfl_xor(s, off, 32);
    float inv = 1.0f / s;
    long rowoff = (long)(rb + r) * 256;
#pragma unroll
    for (int nt = 0; nt < 16; ++nt)
      attnb[rowoff + nt * 16 + l15] = (bf16_t)(p[nt] * inv);
  }
}

// ---------------------------------------------------------------------------
// FFN2 + double LayerNorm + residuals, LDS-staged full-width rows.
// Workgroup tile: 32 rows x 256 cols (full E so LN stays local). 8 waves
// as 2x4; wave tile 16x64 (4 ntiles), K = 1024 (32 k-steps, 4 wmma each).
// ---------------------------------------------------------------------------
__global__ void ffn2_ln_kernel(const bf16_t* hb, const bf16_t* W2T, const float* b2,
                               const float* yf, const float* g1, const float* be1,
                               const float* g2, const float* be2, float* out) {
  __shared__ float F[32][260];  // padded stride vs 64 banks
  __shared__ float mu[32], rs[32];

  const int lane = threadIdx.x & 31;
  const int wv = threadIdx.x >> 5;
  const int wr = wv >> 2, wc = wv & 3;  // 2 x 4 wave grid
  const int lRow = wr * 16;
  const int rowBase = blockIdx.x * 32 + lRow;
  const int colBase = wc * 64;
  const int l15 = lane & 15;
  const int kOff = (lane >> 4) << 3;

  const bf16_t* aPtr = hb + (long)(rowBase + l15) * 1024 + kOff;

  v8f acc[4];
#pragma unroll
  for (int nt = 0; nt < 4; ++nt) acc[nt] = zero8();

  for (int kc = 0; kc < 1024; kc += 32) {
    __builtin_prefetch(aPtr + kc + 128, 0, 1);
    v16bf a = ldA16(aPtr + kc);
#pragma unroll
    for (int nt = 0; nt < 4; ++nt) {
      const bf16_t* bp = W2T + (long)(colBase + nt * 16 + l15) * 1024 + kOff + kc;
      acc[nt] = wmma_bf16(a, ldA16(bp), acc[nt]);
    }
  }

  const int rb = lRow + ((lane >> 4) << 3);
#pragma unroll
  for (int nt = 0; nt < 4; ++nt) {
    int col = colBase + nt * 16 + l15;
    float bv = b2[col];
#pragma unroll
    for (int r = 0; r < 8; ++r) F[rb + r][col] = acc[nt][r] + bv;
  }
  __syncthreads();

  if (threadIdx.x < 32) {  // LN #1 stats
    int row = threadIdx.x;
    float s = 0.f;
    for (int c = 0; c < 256; ++c) s += F[row][c];
    float m = s * (1.f / 256.f);
    float v = 0.f;
    for (int c = 0; c < 256; ++c) { float d = F[row][c] - m; v += d * d; }
    mu[row] = m;
    rs[row] = rsqrtf(v * (1.f / 256.f) + 1e-5f);
  }
  __syncthreads();

  const int c = threadIdx.x;  // one column per thread, sweep 32 rows
  const long gRow0 = (long)blockIdx.x * 32;
  const float gc1 = g1[c], bc1 = be1[c];
  for (int row = 0; row < 32; ++row) {
    float lnv = (F[row][c] - mu[row]) * rs[row] * gc1 + bc1;
    float y2 = yf[(gRow0 + row) * 256 + c] + lnv;  // y + LN(ffn)
    F[row][c] = y2;
  }
  __syncthreads();

  if (threadIdx.x < 32) {  // LN #2 stats on y2
    int row = threadIdx.x;
    float s = 0.f;
    for (int cc = 0; cc < 256; ++cc) s += F[row][cc];
    float m = s * (1.f / 256.f);
    float v = 0.f;
    for (int cc = 0; cc < 256; ++cc) { float d = F[row][cc] - m; v += d * d; }
    mu[row] = m;
    rs[row] = rsqrtf(v * (1.f / 256.f) + 1e-5f);
  }
  __syncthreads();

  const float gc2 = g2[c], bc2 = be2[c];
  for (int row = 0; row < 32; ++row) {
    float y2 = F[row][c];
    out[(gRow0 + row) * 256 + c] = y2 + (y2 - mu[row]) * rs[row] * gc2 + bc2;
  }
}

// ---------------------------------------------------------------------------
// Host pipeline (graph-capture safe: only kernel launches on `stream`).
// Workspace (~640 MB) with dead-buffer aliasing:
//   tb -> attn, qb -> o, kb -> y_bf16.
// ---------------------------------------------------------------------------
extern "C" void kernel_launch(void* const* d_in, const int* in_sizes, int n_in,
                              void* d_out, int out_size, void* d_ws, size_t ws_size,
                              hipStream_t stream) {
  (void)in_sizes; (void)n_in; (void)out_size; (void)ws_size;
  const float* x   = (const float*)d_in[0];
  const float* Wq  = (const float*)d_in[1];
  const float* bq  = (const float*)d_in[2];
  const float* Wk  = (const float*)d_in[3];
  const float* bk  = (const float*)d_in[4];
  const float* Wv  = (const float*)d_in[5];
  const float* bv  = (const float*)d_in[6];
  const float* Wo  = (const float*)d_in[7];
  const float* bo  = (const float*)d_in[8];
  const float* W1  = (const float*)d_in[9];
  const float* b1  = (const float*)d_in[10];
  const float* W2  = (const float*)d_in[11];
  const float* b2  = (const float*)d_in[12];
  const float* g1  = (const float*)d_in[13];
  const float* be1 = (const float*)d_in[14];
  const float* g2  = (const float*)d_in[15];
  const float* be2 = (const float*)d_in[16];
  float* out = (float*)d_out;

  const long NT = 131072;  // B * nW * T = total tokens
  char* ws = (char*)d_ws;
  size_t off = 0;
  auto carve = [&](size_t bytes) -> char* {
    char* p = ws + off;
    off += (bytes + 255) & ~(size_t)255;
    return p;
  };
  bf16_t* WqT = (bf16_t*)carve(65536 * 2);
  bf16_t* WkT = (bf16_t*)carve(65536 * 2);
  bf16_t* WvT = (bf16_t*)carve(65536 * 2);
  bf16_t* WoT = (bf16_t*)carve(65536 * 2);
  bf16_t* W1T = (bf16_t*)carve(262144 * 2);
  bf16_t* W2T = (bf16_t*)carve(262144 * 2);
  bf16_t* tb  = (bf16_t*)carve(NT * 256 * 2);   // later reused as attn
  bf16_t* qb  = (bf16_t*)carve(NT * 256 * 2);   // later reused as O
  bf16_t* kb  = (bf16_t*)carve(NT * 256 * 2);   // later reused as y_bf16
  bf16_t* vtb = (bf16_t*)carve(NT * 256 * 2);   // V^T per window
  float*  yf  = (float*)carve(NT * 256 * 4);    // y (f32) for residuals
  bf16_t* hb  = (bf16_t*)carve(NT * 1024 * 2);  // FFN hidden
  bf16_t* attnb = tb;
  bf16_t* ob    = qb;
  bf16_t* yb    = kb;

  prep_weights_kernel<<<1024, 256, 0, stream>>>(Wq, Wk, Wv, Wo, W1, W2,
                                                WqT, WkT, WvT, WoT, W1T, W2T);
  gather_x_kernel<<<131072, 256, 0, stream>>>(x, tb);

  dim3 gE(2048, 4);  // 131072/64 x 256/64
  gemm_kernel<<<gE, 256, 0, stream>>>(tb, WqT, bq, qb, nullptr, 256, 256, 0);
  gemm_kernel<<<gE, 256, 0, stream>>>(tb, WkT, bk, kb, nullptr, 256, 256, 0);
  gemm_kernel<<<gE, 256, 0, stream>>>(tb, WvT, bv, vtb, nullptr, 256, 256, 1);

  attn_softmax_kernel<<<1024, 256, 0, stream>>>(qb, kb, attnb);

  gemm_kernel<<<gE, 256, 0, stream>>>(attnb, vtb, nullptr, ob, nullptr, 256, 256, 4);
  gemm_kernel<<<gE, 256, 0, stream>>>(ob, WoT, bo, yf, yb, 256, 256, 2);

  dim3 gFF(2048, 16);  // N = 1024
  gemm_kernel<<<gFF, 256, 0, stream>>>(yb, W1T, b1, hb, nullptr, 256, 1024, 3);

  ffn2_ln_kernel<<<4096, 256, 0, stream>>>(hb, W2T, b2, yf, g1, be1, g2, be2, out);
}
